// BiMamba_75479755260445
// MI455X (gfx1250) — compile-verified, matrix operand order checked by reading, not run on hardware
//
#include <hip/hip_runtime.h>
#include <hip/hip_bf16.h>
#include <math.h>

// ---- problem constants ----
#define D_MODEL   768
#define D_INNER   1536
#define D_STATE   128
#define HEADDIM   64
#define NHEADS    24
#define D_CONV    4
#define CONV_DIM  1792          // D_INNER + 2*D_STATE
#define D_IN_PROJ 3352          // 2*D_INNER + 2*D_STATE + NHEADS
#define N_IN_PAD  3392          // padded to multiple of 64 for 16x64 wave tiles
#define BATCH     2
#define SEQLEN    2048
#define ROWS      (BATCH*SEQLEN)   // 4096
#define KMAX      1536
#define TCHUNK    16

typedef __bf16 bf16_t;
typedef __bf16 v16bf __attribute__((ext_vector_type(16)));
typedef float  v8f   __attribute__((ext_vector_type(8)));
typedef unsigned int u32x4 __attribute__((ext_vector_type(4)));
typedef unsigned int u32x4t __attribute__((ext_vector_type(4)));
typedef int          i32x4t __attribute__((ext_vector_type(4)));
typedef int          i32x8t __attribute__((ext_vector_type(8)));

union Frag { v16bf v; u32x4 q[2]; };

// ---------------- conversion kernels ----------------
__global__ void k_f32_to_bf16(const float* __restrict__ src, bf16_t* __restrict__ dst, int n) {
  int i = blockIdx.x * 256 + threadIdx.x;
  if (i < n) dst[i] = (bf16_t)src[i];
}

// in_proj weights: [2][D_IN_PROJ][D_MODEL] f32 -> [2][N_IN_PAD][D_MODEL] bf16 (zero pad rows)
__global__ void k_cvt_w_in(const float* __restrict__ w, bf16_t* __restrict__ dst) {
  int i = blockIdx.x * 256 + threadIdx.x;
  int total = 2 * N_IN_PAD * D_MODEL;
  if (i >= total) return;
  int c = i % D_MODEL;
  int r = (i / D_MODEL) % N_IN_PAD;
  int d = i / (D_MODEL * N_IN_PAD);
  float v = (r < D_IN_PROJ) ? w[((size_t)d * D_IN_PROJ + r) * D_MODEL + c] : 0.0f;
  dst[i] = (bf16_t)v;
}

// ---------------- WMMA bf16 GEMM ----------------
// C[M=4096, Npad] (f32, ld=Npad) = A[M,K]bf16 @ Bw[Npad,K]bf16^T
// block: 256 threads = 8 waves; block tile 16(M) x 512(N); wave tile 16x64
// (4 accumulators -> A fragment reused across 4 WMMAs).
// A tile (16 x K) staged to LDS by the Tensor Data Mover when available
// (forward direction); backward direction gathers rows time-reversed manually.
__global__ __launch_bounds__(256)
void k_gemm_bf16(const bf16_t* __restrict__ A, const bf16_t* __restrict__ Bw,
                 float* __restrict__ C, int K, int Npad, int flip) {
  __shared__ __align__(16) bf16_t aT[16 * KMAX];
  int tid = threadIdx.x;
  int lane = tid & 31, wid = tid >> 5;
  int m0 = blockIdx.y * 16;

  bool staged = false;
#if __has_builtin(__builtin_amdgcn_tensor_load_to_lds) && __has_builtin(__builtin_amdgcn_s_wait_tensorcnt)
  if (!flip) {
    staged = true;
    if (wid == 0) {
      // Build Tensor DMA Descriptor (D#) per CDNA5 ISA ch.8:
      // 2D tensor, data_size=2B, tile = 16 rows x K cols, row stride = K.
      unsigned ldsAddr = (unsigned)(size_t)&aT[0];       // addr[31:0] = LDS offset
      unsigned long long ga = (unsigned long long)(size_t)(A + (size_t)m0 * K);
      u32x4t g0;
      g0[0] = 1u;                                        // count=1 (valid), user mode
      g0[1] = ldsAddr;                                   // lds_addr
      g0[2] = (unsigned)(ga & 0xFFFFFFFFu);              // global_addr[31:0]
      g0[3] = (unsigned)((ga >> 32) & 0x01FFFFFFu) | (2u << 30); // ga[56:32], type=2
      i32x8t g1;
      g1[0] = (int)(1u << 16);                           // data_size=1 -> 2 bytes
      g1[1] = (int)(((unsigned)K & 0xFFFFu) << 16);      // tensor_dim0[15:0]
      g1[2] = (int)((((unsigned)K >> 16) & 0xFFFFu) | (16u << 16)); // dim0[31:16], tensor_dim1=16
      g1[3] = (int)((unsigned)K << 16);                  // tile_dim0 = K
      g1[4] = 16;                                        // tile_dim1 = 16 rows
      g1[5] = K;                                         // tensor_dim0_stride[31:0]
      g1[6] = 0;
      g1[7] = 0;
      i32x4t z4 = {};
#if defined(__clang_major__) && (__clang_major__ >= 23)
      i32x8t z8 = {};
      __builtin_amdgcn_tensor_load_to_lds(g0, g1, z4, z4, z8, 0);
#else
      __builtin_amdgcn_tensor_load_to_lds(g0, g1, z4, z4, 0);
#endif
      __builtin_amdgcn_s_wait_tensorcnt(0);              // TENSORcnt is per-wave
    }
  }
#endif
  if (!staged) {
    int chunks = (16 * K) / 8;
    for (int i = tid; i < chunks; i += 256) {
      int row = (i * 8) / K;
      int col = (i * 8) % K;
      int mg  = m0 + row;
      int b   = mg / SEQLEN, t = mg % SEQLEN;
      int srow = flip ? (b * SEQLEN + (SEQLEN - 1 - t)) : mg;
      *(u32x4*)&aT[row * K + col] = *(const u32x4*)&A[(size_t)srow * K + col];
    }
  }
  __syncthreads();

  int n0 = blockIdx.x * 512 + wid * 64;
  if (n0 >= Npad) return;          // wave-uniform (Npad multiple of 64)

  int mr = lane & 15;              // A row / B row within 16-tile
  int kh = (lane >> 4) * 8;        // K-half select per ISA 16-bit A/B layout

  const bf16_t* abase = aT + mr * K + kh;
  const bf16_t* bb0 = Bw + (size_t)(n0 +  0 + mr) * K + kh;
  const bf16_t* bb1 = Bw + (size_t)(n0 + 16 + mr) * K + kh;
  const bf16_t* bb2 = Bw + (size_t)(n0 + 32 + mr) * K + kh;
  const bf16_t* bb3 = Bw + (size_t)(n0 + 48 + mr) * K + kh;

  v8f acc0 = {}, acc1 = {}, acc2 = {}, acc3 = {};
  for (int kb = 0; kb < K; kb += 32) {
    Frag a;
    a.q[0] = *(const u32x4*)(abase + kb);        // K = kh+0..7   -> VGPR0-3
    a.q[1] = *(const u32x4*)(abase + kb + 16);   // K = kh+16..23 -> VGPR4-7
    __builtin_prefetch(bb0 + kb + 128, 0, 1);    // global_prefetch next chunk
    Frag b;
    b.q[0] = *(const u32x4*)(bb0 + kb);
    b.q[1] = *(const u32x4*)(bb0 + kb + 16);
    acc0 = __builtin_amdgcn_wmma_f32_16x16x32_bf16(false, a.v, false, b.v, (short)0, acc0, false, false);
    b.q[0] = *(const u32x4*)(bb1 + kb);
    b.q[1] = *(const u32x4*)(bb1 + kb + 16);
    acc1 = __builtin_amdgcn_wmma_f32_16x16x32_bf16(false, a.v, false, b.v, (short)0, acc1, false, false);
    b.q[0] = *(const u32x4*)(bb2 + kb);
    b.q[1] = *(const u32x4*)(bb2 + kb + 16);
    acc2 = __builtin_amdgcn_wmma_f32_16x16x32_bf16(false, a.v, false, b.v, (short)0, acc2, false, false);
    b.q[0] = *(const u32x4*)(bb3 + kb);
    b.q[1] = *(const u32x4*)(bb3 + kb + 16);
    acc3 = __builtin_amdgcn_wmma_f32_16x16x32_bf16(false, a.v, false, b.v, (short)0, acc3, false, false);
  }

  int crow = m0 + ((lane >> 4) << 3);   // lanes 16-31 hold M = 8+r
  int ccl  = lane & 15;
  #pragma unroll
  for (int r = 0; r < 8; ++r) {
    C[(size_t)(crow + r) * Npad + n0 +  0 + ccl] = acc0[r];
    C[(size_t)(crow + r) * Npad + n0 + 16 + ccl] = acc1[r];
    C[(size_t)(crow + r) * Npad + n0 + 32 + ccl] = acc2[r];
    C[(size_t)(crow + r) * Npad + n0 + 48 + ccl] = acc3[r];
  }
}

// ---------------- depthwise conv(4) + SiLU ----------------
__global__ void k_conv(const float* __restrict__ zx, const float* __restrict__ conv_w,
                       const float* __restrict__ conv_b, float* __restrict__ xconv) {
  int i = blockIdx.x * 256 + threadIdx.x;
  int dir = blockIdx.y;
  if (i >= ROWS * CONV_DIM) return;
  int c = i % CONV_DIM;
  int row = i / CONV_DIM;
  int b = row / SEQLEN, t = row % SEQLEN;
  const float* zxd = zx + (size_t)dir * ROWS * N_IN_PAD;
  float acc = conv_b[dir * CONV_DIM + c];
  #pragma unroll
  for (int k = 0; k < D_CONV; ++k) {
    int ts = t - (D_CONV - 1) + k;
    if (ts >= 0)
      acc += zxd[(size_t)(b * SEQLEN + ts) * N_IN_PAD + D_INNER + c] *
             conv_w[(dir * CONV_DIM + c) * D_CONV + k];
  }
  xconv[(size_t)dir * ROWS * CONV_DIM + i] = acc / (1.0f + expf(-acc)); // silu
}

// ---------------- dt = softplus(raw+bias), dA = exp(dt*A) ----------------
__global__ void k_dt(const float* __restrict__ zx, const float* __restrict__ dt_bias,
                     const float* __restrict__ A_log,
                     float* __restrict__ dt_o, float* __restrict__ dA_o) {
  int i = blockIdx.x * 256 + threadIdx.x;
  int dir = blockIdx.y;
  if (i >= ROWS * NHEADS) return;
  int h = i % NHEADS;
  int row = i / NHEADS;
  float raw = zx[(size_t)dir * ROWS * N_IN_PAD + (size_t)row * N_IN_PAD +
                 D_INNER + CONV_DIM + h] + dt_bias[dir * NHEADS + h];
  float dt = (raw > 20.0f) ? raw : log1pf(expf(raw));
  float A = -expf(A_log[dir * NHEADS + h]);
  dt_o[(size_t)dir * ROWS * NHEADS + i] = dt;
  dA_o[(size_t)dir * ROWS * NHEADS + i] = expf(dt * A);
}

// ---------------- selective-state scan ----------------
// one block per (dir, batch, head); state h[64][128] f32 fully register-resident:
// wave w owns rows p = w*8..w*8+7, lane owns columns n = lane + 32*j (j=0..3).
__global__ __launch_bounds__(256)
void k_scan(const float* __restrict__ xconv, const float* __restrict__ dt,
            const float* __restrict__ dA, const float* __restrict__ Dp,
            float* __restrict__ y) {
  __shared__ float sX[TCHUNK][HEADDIM];
  __shared__ float sB[TCHUNK][D_STATE];
  __shared__ float sC[TCHUNK][D_STATE];
  __shared__ float sdA[TCHUNK];
  __shared__ float sdt[TCHUNK];

  int head = blockIdx.x % NHEADS;
  int b    = (blockIdx.x / NHEADS) % BATCH;
  int dir  = blockIdx.x / (NHEADS * BATCH);
  int tid = threadIdx.x, lane = tid & 31, wid = tid >> 5;
  int p0 = wid * 8;

  float Dh = Dp[dir * NHEADS + head];
  const float* xc  = xconv + ((size_t)dir * ROWS + (size_t)b * SEQLEN) * CONV_DIM;
  const float* dtp = dt + ((size_t)dir * ROWS + (size_t)b * SEQLEN) * NHEADS + head;
  const float* dAp = dA + ((size_t)dir * ROWS + (size_t)b * SEQLEN) * NHEADS + head;
  float* yp = y + ((size_t)dir * ROWS + (size_t)b * SEQLEN) * D_INNER + head * HEADDIM;

  float h[8][4];
  #pragma unroll
  for (int p = 0; p < 8; ++p)
    #pragma unroll
    for (int j = 0; j < 4; ++j) h[p][j] = 0.0f;

  for (int t0 = 0; t0 < SEQLEN; t0 += TCHUNK) {
    __syncthreads();
    for (int i = tid; i < TCHUNK * HEADDIM; i += 256) {
      int tt = i / HEADDIM, p = i % HEADDIM;
      sX[tt][p] = xc[(size_t)(t0 + tt) * CONV_DIM + head * HEADDIM + p];
    }
    for (int i = tid; i < TCHUNK * D_STATE; i += 256) {
      int tt = i / D_STATE, n = i % D_STATE;
      sB[tt][n] = xc[(size_t)(t0 + tt) * CONV_DIM + D_INNER + n];
      sC[tt][n] = xc[(size_t)(t0 + tt) * CONV_DIM + D_INNER + D_STATE + n];
    }
    if (tid < TCHUNK) {
      sdt[tid] = dtp[(size_t)(t0 + tid) * NHEADS];
      sdA[tid] = dAp[(size_t)(t0 + tid) * NHEADS];
    }
    __syncthreads();

    for (int tt = 0; tt < TCHUNK; ++tt) {
      float dA_t = sdA[tt], dt_t = sdt[tt];
      float Bn[4], Cn[4];
      #pragma unroll
      for (int j = 0; j < 4; ++j) {
        Bn[j] = sB[tt][j * 32 + lane];
        Cn[j] = sC[tt][j * 32 + lane];
      }
      float yv[8];
      #pragma unroll
      for (int p = 0; p < 8; ++p) {
        float xv = sX[tt][p0 + p];
        float a = dt_t * xv;
        float acc = 0.0f;
        #pragma unroll
        for (int j = 0; j < 4; ++j) {
          h[p][j] = fmaf(h[p][j], dA_t, a * Bn[j]);
          acc = fmaf(h[p][j], Cn[j], acc);
        }
        #pragma unroll
        for (int off = 16; off >= 1; off >>= 1) acc += __shfl_xor(acc, off, 32);
        yv[p] = acc + Dh * xv;
      }
      if (lane == 0) {
        #pragma unroll
        for (int p = 0; p < 8; ++p)
          yp[(size_t)(t0 + tt) * D_INNER + p0 + p] = yv[p];
      }
    }
  }
}

// ---------------- gated RMSNorm -> bf16 ----------------
__global__ __launch_bounds__(256)
void k_gatenorm(const float* __restrict__ y, const float* __restrict__ zx,
                const float* __restrict__ norm_w, bf16_t* __restrict__ out) {
  int row = blockIdx.x;               // dir*ROWS + (b*L+t)
  int dir = row / ROWS;
  int tid = threadIdx.x, lane = tid & 31, wid = tid >> 5;
  const float* yr = y + (size_t)row * D_INNER;
  const float* zr = zx + (size_t)row * N_IN_PAD;

  float u[6];
  float s = 0.0f;
  #pragma unroll
  for (int i = 0; i < 6; ++i) {
    int c = tid + i * 256;
    float z = zr[c];
    float uv = yr[c] * (z / (1.0f + expf(-z)));
    u[i] = uv;
    s += uv * uv;
  }
  #pragma unroll
  for (int off = 16; off >= 1; off >>= 1) s += __shfl_xor(s, off, 32);
  __shared__ float red[8];
  if (lane == 0) red[wid] = s;
  __syncthreads();
  float tot = 0.0f;
  #pragma unroll
  for (int w = 0; w < 8; ++w) tot += red[w];
  float scale = rsqrtf(tot / (float)D_INNER + 1e-5f);
  #pragma unroll
  for (int i = 0; i < 6; ++i) {
    int c = tid + i * 256;
    out[(size_t)row * D_INNER + c] = (bf16_t)(u[i] * scale * norm_w[dir * D_INNER + c]);
  }
}

// ---------------- final: out = proj_fwd + flip(proj_bwd_rev) ----------------
__global__ void k_final(const float* __restrict__ proj, float* __restrict__ out) {
  int i = blockIdx.x * 256 + threadIdx.x;
  if (i >= ROWS * D_MODEL) return;
  int c = i % D_MODEL;
  int row = i / D_MODEL;
  int b = row / SEQLEN, t = row % SEQLEN;
  int rflip = b * SEQLEN + (SEQLEN - 1 - t);
  out[i] = proj[i] + proj[(size_t)ROWS * D_MODEL + (size_t)rflip * D_MODEL + c];
}

// ---------------- host orchestration ----------------
extern "C" void kernel_launch(void* const* d_in, const int* in_sizes, int n_in,
                              void* d_out, int out_size, void* d_ws, size_t ws_size,
                              hipStream_t stream) {
  (void)in_sizes; (void)n_in; (void)out_size; (void)ws_size;
  const float* x       = (const float*)d_in[0];
  const float* in_w    = (const float*)d_in[1];
  const float* conv_w  = (const float*)d_in[2];
  const float* conv_b  = (const float*)d_in[3];
  const float* dt_bias = (const float*)d_in[4];
  const float* A_log   = (const float*)d_in[5];
  const float* Dp      = (const float*)d_in[6];
  const float* norm_w  = (const float*)d_in[7];
  const float* out_w   = (const float*)d_in[8];

  char* ws = (char*)d_ws;
  size_t off = 0;
  auto alloc = [&](size_t bytes) -> void* {
    void* p = ws + off;
    off += (bytes + 255) & ~(size_t)255;
    return p;
  };
  bf16_t* x_bf    = (bf16_t*)alloc((size_t)ROWS * D_MODEL * 2);
  bf16_t* win_bf  = (bf16_t*)alloc((size_t)2 * N_IN_PAD * D_MODEL * 2);
  bf16_t* wout_bf = (bf16_t*)alloc((size_t)2 * D_MODEL * D_INNER * 2);
  float*  zx      = (float*) alloc((size_t)2 * ROWS * N_IN_PAD * 4);
  float*  xconv   = (float*) alloc((size_t)2 * ROWS * CONV_DIM * 4);
  float*  dtb     = (float*) alloc((size_t)2 * ROWS * NHEADS * 4);
  float*  dAb     = (float*) alloc((size_t)2 * ROWS * NHEADS * 4);
  float*  yscan   = (float*) alloc((size_t)2 * ROWS * D_INNER * 4);
  bf16_t* ynorm   = (bf16_t*)alloc((size_t)2 * ROWS * D_INNER * 2);
  float*  proj    = (float*) alloc((size_t)2 * ROWS * D_MODEL * 4);

  // precision conversions
  k_f32_to_bf16<<<(ROWS * D_MODEL + 255) / 256, 256, 0, stream>>>(x, x_bf, ROWS * D_MODEL);
  k_cvt_w_in<<<(2 * N_IN_PAD * D_MODEL + 255) / 256, 256, 0, stream>>>(in_w, win_bf);
  k_f32_to_bf16<<<(2 * D_MODEL * D_INNER + 255) / 256, 256, 0, stream>>>(
      out_w, wout_bf, 2 * D_MODEL * D_INNER);

  // in_proj GEMMs (fwd, and bwd with time-flip folded into the A gather)
  dim3 gin((N_IN_PAD + 511) / 512, ROWS / 16);
  k_gemm_bf16<<<gin, 256, 0, stream>>>(x_bf, win_bf, zx, D_MODEL, N_IN_PAD, 0);
  k_gemm_bf16<<<gin, 256, 0, stream>>>(x_bf, win_bf + (size_t)N_IN_PAD * D_MODEL,
                                       zx + (size_t)ROWS * N_IN_PAD, D_MODEL, N_IN_PAD, 1);

  dim3 gconv((ROWS * CONV_DIM + 255) / 256, 2);
  k_conv<<<gconv, 256, 0, stream>>>(zx, conv_w, conv_b, xconv);
  dim3 gdt((ROWS * NHEADS + 255) / 256, 2);
  k_dt<<<gdt, 256, 0, stream>>>(zx, dt_bias, A_log, dtb, dAb);

  // selective scan: 96 independent (dir,batch,head) blocks
  k_scan<<<2 * BATCH * NHEADS, 256, 0, stream>>>(xconv, dtb, dAb, Dp, yscan);

  // gated RMSNorm -> bf16 activations for out_proj
  k_gatenorm<<<2 * ROWS, 256, 0, stream>>>(yscan, zx, norm_w, ynorm);

  // out_proj GEMMs
  dim3 gout((D_MODEL + 511) / 512, ROWS / 16);
  k_gemm_bf16<<<gout, 256, 0, stream>>>(ynorm, wout_bf, proj, D_INNER, D_MODEL, 0);
  k_gemm_bf16<<<gout, 256, 0, stream>>>(ynorm + (size_t)ROWS * D_INNER,
                                        wout_bf + (size_t)D_MODEL * D_INNER,
                                        proj + (size_t)ROWS * D_MODEL, D_INNER, D_MODEL, 0);

  // combine fwd + flipped bwd
  k_final<<<(ROWS * D_MODEL + 255) / 256, 256, 0, stream>>>(proj, (float*)d_out);
}
